// HypothesisGAT_3959959847189
// MI455X (gfx1250) — matrix operand order repaired and checked by reading, not compiled, not run on hardware
//
#include <hip/hip_runtime.h>
#include <hip/hip_bf16.h>

// ---------------------------------------------------------------------------
// Problem constants (match reference)
// ---------------------------------------------------------------------------
#define NNODES 50000
#define NEDG   400000
#define DIN    6
#define HID    128
#define NH     4
#define NL     3
#define NHYP   4096
#define KDIM   128          // every GEMM in this net has K = 128

typedef __bf16 bf16;
typedef bf16 v16bf __attribute__((ext_vector_type(16)));
typedef unsigned short v16us __attribute__((ext_vector_type(16)));
typedef unsigned short v8us  __attribute__((ext_vector_type(8)));
typedef float v8f __attribute__((ext_vector_type(8)));

// ---------------------------------------------------------------------------
// Helpers
// ---------------------------------------------------------------------------
__device__ __forceinline__ unsigned short f2bf(float f) {
    unsigned int u = __float_as_uint(f);
    unsigned int r = u + 0x7FFFu + ((u >> 16) & 1u);   // round-to-nearest-even
    return (unsigned short)(r >> 16);
}

// order-preserving float <-> uint transform (atomic float max via uint max)
__device__ __forceinline__ unsigned int f2ord(float f) {
    unsigned int u = __float_as_uint(f);
    return (u & 0x80000000u) ? ~u : (u | 0x80000000u);
}
__device__ __forceinline__ float ord2f(unsigned int u) {
    unsigned int v = (u & 0x80000000u) ? (u & 0x7FFFFFFFu) : ~u;
    return __uint_as_float(v);
}

// ---------------------------------------------------------------------------
// f32 -> bf16 conversion (activations)
// ---------------------------------------------------------------------------
__global__ void cvt_bf16_k(const float* __restrict__ in, unsigned short* __restrict__ out,
                           long n) {
    long t = (long)blockIdx.x * blockDim.x + threadIdx.x;
    if (t < n) out[t] = f2bf(in[t]);
}

// ---------------------------------------------------------------------------
// Pack a row-major f32 weight matrix W[K, Nc] into WMMA B-fragment-major bf16:
// for N-tile nt, K-tile kt (32 deep), lane l (half = l>>4, r = l&15), halves i:
//   out[(((nt*(K/32) + kt)*32 + l)*16 + i] = bf16(W[kt*32 + half*16 + i, nt*16 + r])
// After packing, each lane's B fragment is 32 contiguous bytes (2 x b128 loads).
// ---------------------------------------------------------------------------
__global__ void pack_b_k(const float* __restrict__ W, unsigned short* __restrict__ out,
                         int K, int Nc) {
    int t = blockIdx.x * blockDim.x + threadIdx.x;
    if (t >= K * Nc) return;
    int i    = t & 15;
    int lane = (t >> 4) & 31;
    int t2   = t >> 9;                 // (kt, nt) tile index; 512 elems per tile
    int ktn  = K >> 5;
    int kt   = t2 % ktn;
    int nt   = t2 / ktn;
    int half = lane >> 4, r = lane & 15;
    int krow = kt * 32 + half * 16 + i;
    int col  = nt * 16 + r;
    out[t] = f2bf(W[(size_t)krow * Nc + col]);
}

// ---------------------------------------------------------------------------
// Encoder layer 1: t = relu(x @ W1 + b1), x:[N,6] W1:[6,128] -> bf16 [N,128]
// ---------------------------------------------------------------------------
__global__ void enc1_k(const float* __restrict__ x, const float* __restrict__ W1,
                       const float* __restrict__ b1, unsigned short* __restrict__ out) {
    int t = blockIdx.x * blockDim.x + threadIdx.x;
    if (t >= NNODES * HID) return;
    int n = t >> 7, j = t & 127;
    float s = b1[j];
#pragma unroll
    for (int k = 0; k < DIN; ++k) s += x[n * DIN + k] * W1[k * HID + j];
    out[t] = f2bf(fmaxf(s, 0.f));
}

// ---------------------------------------------------------------------------
// WMMA bf16 GEMM, K fixed to 128: C[M,Nc] = A[M,128] @ B + bias (+relu).
// A: row-major bf16.  Bp: fragment-packed (pack_b_k).  One wave computes a
// 16(M) x 64(N) strip = 4 N-tiles reusing one A fragment; 16 unrolled WMMAs.
// Requires: M % 16 == 0, Nc % 64 == 0.
// ---------------------------------------------------------------------------
__global__ void wmma_gemm_bf16_k(const unsigned short* __restrict__ A,
                                 const unsigned short* __restrict__ Bp,
                                 const float* __restrict__ bias,
                                 float* __restrict__ C,
                                 int M, int Nc, int relu) {
    int gtid = blockIdx.x * blockDim.x + threadIdx.x;
    int wave = gtid >> 5;
    int lane = threadIdx.x & 31;
    int nGroups = Nc >> 6;                     // groups of 4 N-tiles
    int tileM = (wave / nGroups) << 4;
    int ng    = wave % nGroups;
    if (tileM >= M) return;

    int half = lane >> 4;                      // 0: lanes 0-15, 1: lanes 16-31
    int r    = lane & 15;

    const unsigned short* Ap = A + (size_t)(tileM + r) * KDIM;

    v8f acc[4] = {v8f{}, v8f{}, v8f{}, v8f{}};

#pragma unroll
    for (int kt = 0; kt < KDIM / 32; ++kt) {
        const int k0 = kt * 32;
        // A fragment (matches packed B): half 0 -> K=k0+[0..15];
        //                                half 1 -> K=k0+16+[0..15].
        v8us alo = *(const v8us*)(Ap + k0 + half * 16);
        v8us ahi = *(const v8us*)(Ap + k0 + half * 16 + 8);
        v16us au;
#pragma unroll
        for (int i = 0; i < 8; ++i) { au[i] = alo[i]; au[8 + i] = ahi[i]; }
        v16bf a = __builtin_bit_cast(v16bf, au);

#pragma unroll
        for (int t = 0; t < 4; ++t) {
            int nt = ng * 4 + t;
            const unsigned short* bp =
                Bp + ((size_t)(nt * (KDIM / 32) + kt) * 32 + lane) * 16;
            v16bf b = __builtin_bit_cast(v16bf, *(const v16us*)bp);
            acc[t] = __builtin_amdgcn_wmma_f32_16x16x32_bf16(
                false, a, false, b, (short)0, acc[t], false, false);
        }
    }

    // D layout: vgpr j holds (M = tileM + 8*half + j, N = tileN + r)
    int row0 = tileM + half * 8;
    int colBase = ng * 64 + r;
    float bv[4];
#pragma unroll
    for (int t = 0; t < 4; ++t) bv[t] = bias ? bias[colBase + t * 16] : 0.f;

#pragma unroll
    for (int j = 0; j < 8; ++j) {
        float* p = C + (size_t)(row0 + j) * Nc + colBase;
#pragma unroll
        for (int t = 0; t < 4; ++t) {
            float v = acc[t][j] + bv[t];
            if (relu) v = fmaxf(v, 0.f);
            p[t * 16] = v;
        }
    }
}

// ---------------------------------------------------------------------------
// Per-node attention projections: a_s[n,h] = <hW[n,h,:], att_src[h,:]>, ditto dst
// ---------------------------------------------------------------------------
__global__ void attn_scores_k(const float* __restrict__ hW,
                              const float* __restrict__ att_src,
                              const float* __restrict__ att_dst,
                              float* __restrict__ a_s, float* __restrict__ a_d) {
    int t = blockIdx.x * blockDim.x + threadIdx.x;   // over N*H
    if (t >= NNODES * NH) return;
    int n = t >> 2, h = t & 3;
    const float* row = hW + (size_t)n * (NH * HID) + h * HID;
    const float* as  = att_src + h * HID;
    const float* ad  = att_dst + h * HID;
    float s = 0.f, d = 0.f;
#pragma unroll 8
    for (int c = 0; c < HID; ++c) { float v = row[c]; s += v * as[c]; d += v * ad[c]; }
    a_s[t] = s; a_d[t] = d;
}

// zero the atomic buffers for a layer
__global__ void init_layer_k(float* __restrict__ accum, unsigned int* __restrict__ m_u,
                             float* __restrict__ denom) {
    int t = blockIdx.x * blockDim.x + threadIdx.x;
    if (t < NNODES * HID) accum[t] = 0.f;
    if (t < NNODES * NH) { m_u[t] = 0u; denom[t] = 0.f; }
}

__device__ __forceinline__ void edge_ends(const int* ei, int e, int& src, int& dst) {
    if (e < NEDG) { src = ei[e]; dst = ei[NEDG + e]; }
    else          { src = e - NEDG; dst = e - NEDG; }           // appended self-loops
}

// pass 1: segment max of leaky_relu(a_s[src]+a_d[dst]) over dst
__global__ void edge_max_k(const int* __restrict__ ei, const float* __restrict__ a_s,
                           const float* __restrict__ a_d, unsigned int* __restrict__ m_u) {
    int e = blockIdx.x * blockDim.x + threadIdx.x;
    if (e >= NEDG + NNODES) return;
    int src, dst; edge_ends(ei, e, src, dst);
#pragma unroll
    for (int h = 0; h < NH; ++h) {
        float v = a_s[src * NH + h] + a_d[dst * NH + h];
        v = v > 0.f ? v : 0.2f * v;
        atomicMax(&m_u[dst * NH + h], f2ord(v));
    }
}

// pass 2: denom[dst,h] += exp(e - m[dst,h])
__global__ void edge_sum_k(const int* __restrict__ ei, const float* __restrict__ a_s,
                           const float* __restrict__ a_d, const unsigned int* __restrict__ m_u,
                           float* __restrict__ denom) {
    int e = blockIdx.x * blockDim.x + threadIdx.x;
    if (e >= NEDG + NNODES) return;
    int src, dst; edge_ends(ei, e, src, dst);
#pragma unroll
    for (int h = 0; h < NH; ++h) {
        float v = a_s[src * NH + h] + a_d[dst * NH + h];
        v = v > 0.f ? v : 0.2f * v;
        atomicAdd(&denom[dst * NH + h], __expf(v - ord2f(m_u[dst * NH + h])));
    }
}

// pass 3: accum[dst,c] += (1/H) * sum_h alpha[e,h] * hW[src,h,c]
__global__ void edge_msg_k(const int* __restrict__ ei, const float* __restrict__ a_s,
                           const float* __restrict__ a_d, const unsigned int* __restrict__ m_u,
                           const float* __restrict__ denom, const float* __restrict__ hW,
                           float* __restrict__ accum) {
    __shared__ float alpha[NH];
    int e = blockIdx.x;
    if (e >= NEDG + NNODES) return;
    int src, dst; edge_ends(ei, e, src, dst);
    int c = threadIdx.x;                                   // 0..127
    if (c < NH) {
        float v = a_s[src * NH + c] + a_d[dst * NH + c];
        v = v > 0.f ? v : 0.2f * v;
        float ex = __expf(v - ord2f(m_u[dst * NH + c]));
        alpha[c] = ex / (denom[dst * NH + c] + 1e-16f);
    }
    __syncthreads();
    const float* hs = hW + (size_t)src * (NH * HID);
    float v = 0.f;
#pragma unroll
    for (int h = 0; h < NH; ++h) v += alpha[h] * hs[h * HID + c];
    atomicAdd(&accum[(size_t)dst * HID + c], v * 0.25f);   // mean over 4 heads
}

// bias + LayerNorm + ReLU + residual; writes f32 h and bf16 hb
__global__ void finalize_k(const float* __restrict__ accum, const float* __restrict__ gat_b,
                           const float* __restrict__ ln_g, const float* __restrict__ ln_b,
                           float* __restrict__ h, unsigned short* __restrict__ hb) {
    __shared__ float red[HID];
    int n = blockIdx.x, c = threadIdx.x;
    float v = accum[(size_t)n * HID + c] + gat_b[c];
    red[c] = v; __syncthreads();
    for (int s = HID / 2; s > 0; s >>= 1) { if (c < s) red[c] += red[c + s]; __syncthreads(); }
    float mu = red[0] * (1.f / HID); __syncthreads();
    float d = v - mu;
    red[c] = d * d; __syncthreads();
    for (int s = HID / 2; s > 0; s >>= 1) { if (c < s) red[c] += red[c + s]; __syncthreads(); }
    float var = red[0] * (1.f / HID);
    float ln = d * rsqrtf(var + 1e-5f) * ln_g[c] + ln_b[c];
    ln = fmaxf(ln, 0.f);
    float out = ln + h[(size_t)n * HID + c];               // residual (h still holds input)
    h[(size_t)n * HID + c] = out;
    hb[(size_t)n * HID + c] = f2bf(out);
}

// gather hypothesis rows -> bf16
__global__ void gather_k(const float* __restrict__ h, const int* __restrict__ idx,
                         unsigned short* __restrict__ out) {
    int t = blockIdx.x * blockDim.x + threadIdx.x;
    if (t >= NHYP * HID) return;
    int i = t >> 7, c = t & 127;
    out[t] = f2bf(h[(size_t)idx[i] * HID + c]);
}

// final head: logits[i] = <hyp_t[i,:64], W2> + b2
__global__ void head2_k(const float* __restrict__ hyp_t, const float* __restrict__ W2,
                        const float* __restrict__ b2, float* __restrict__ out) {
    int i = blockIdx.x * blockDim.x + threadIdx.x;
    if (i >= NHYP) return;
    float s = b2[0];
#pragma unroll 8
    for (int c = 0; c < 64; ++c) s += hyp_t[(size_t)i * 64 + c] * W2[c];
    out[i] = s;
}

// ---------------------------------------------------------------------------
// Host side
// ---------------------------------------------------------------------------
static inline void launch_gemm(const unsigned short* A, const unsigned short* Bp,
                               const float* bias, float* C, int M, int Nc,
                               int relu, hipStream_t s) {
    long waves = (long)(M / 16) * (Nc / 64);
    long threads = waves * 32;
    int block = 256;
    long grid = (threads + block - 1) / block;
    wmma_gemm_bf16_k<<<(int)grid, block, 0, s>>>(A, Bp, bias, C, M, Nc, relu);
}

extern "C" void kernel_launch(void* const* d_in, const int* in_sizes, int n_in,
                              void* d_out, int out_size, void* d_ws, size_t ws_size,
                              hipStream_t stream) {
    const float* x        = (const float*)d_in[0];
    const int*   ei       = (const int*)d_in[1];
    const int*   hyp_idx  = (const int*)d_in[2];
    const float* enc_W1   = (const float*)d_in[3];
    const float* enc_b1   = (const float*)d_in[4];
    const float* enc_W2   = (const float*)d_in[5];
    const float* enc_b2   = (const float*)d_in[6];
    const float* gat_W    = (const float*)d_in[7];
    const float* att_src  = (const float*)d_in[8];
    const float* att_dst  = (const float*)d_in[9];
    const float* gat_b    = (const float*)d_in[10];
    const float* ln_g     = (const float*)d_in[11];
    const float* ln_b     = (const float*)d_in[12];
    const float* head_W1  = (const float*)d_in[13];
    const float* head_b1  = (const float*)d_in[14];
    const float* head_W2  = (const float*)d_in[15];
    const float* head_b2  = (const float*)d_in[16];
    float* logits = (float*)d_out;

    // workspace carve-out
    char* ws = (char*)d_ws; size_t off = 0;
    auto carve = [&](size_t bytes) -> void* {
        void* p = ws + off; off = (off + bytes + 255) & ~(size_t)255; return p;
    };
    float*          h_    = (float*)carve((size_t)NNODES * HID * 4);
    unsigned short* hb    = (unsigned short*)carve((size_t)NNODES * HID * 2);
    float*          hW    = (float*)carve((size_t)NNODES * NH * HID * 4);
    float*          a_s   = (float*)carve((size_t)NNODES * NH * 4);
    float*          a_d   = (float*)carve((size_t)NNODES * NH * 4);
    unsigned int*   m_u   = (unsigned int*)carve((size_t)NNODES * NH * 4);
    float*          denom = (float*)carve((size_t)NNODES * NH * 4);
    float*          accum = (float*)carve((size_t)NNODES * HID * 4);
    unsigned short* gat_Wp   = (unsigned short*)carve((size_t)NL * KDIM * NH * HID * 2);
    unsigned short* enc_W2p  = (unsigned short*)carve((size_t)KDIM * HID * 2);
    unsigned short* head_W1p = (unsigned short*)carve((size_t)KDIM * 64 * 2);
    unsigned short* hyp_b    = (unsigned short*)carve((size_t)NHYP * HID * 2);
    float*          hyp_t    = (float*)carve((size_t)NHYP * 64 * 4);

    const int BLK = 256;
    auto gsz = [&](long n) { return (int)((n + BLK - 1) / BLK); };

    // pack weights into WMMA fragment-major bf16
    for (int l = 0; l < NL; ++l)
        pack_b_k<<<gsz((long)KDIM * NH * HID), BLK, 0, stream>>>(
            gat_W + (size_t)l * KDIM * NH * HID,
            gat_Wp + (size_t)l * KDIM * NH * HID, KDIM, NH * HID);
    pack_b_k<<<gsz((long)KDIM * HID), BLK, 0, stream>>>(enc_W2, enc_W2p, KDIM, HID);
    pack_b_k<<<gsz((long)KDIM * 64), BLK, 0, stream>>>(head_W1, head_W1p, KDIM, 64);

    // encoder
    enc1_k<<<gsz((long)NNODES * HID), BLK, 0, stream>>>(x, enc_W1, enc_b1, hb);
    launch_gemm(hb, enc_W2p, enc_b2, h_, NNODES, HID, 0, stream);
    cvt_bf16_k<<<gsz((long)NNODES * HID), BLK, 0, stream>>>(h_, hb, (long)NNODES * HID);

    // GAT layers
    const long EP = (long)NEDG + NNODES;
    for (int l = 0; l < NL; ++l) {
        launch_gemm(hb, gat_Wp + (size_t)l * KDIM * NH * HID, nullptr, hW,
                    NNODES, NH * HID, 0, stream);
        attn_scores_k<<<gsz((long)NNODES * NH), BLK, 0, stream>>>(
            hW, att_src + (size_t)l * NH * HID, att_dst + (size_t)l * NH * HID, a_s, a_d);
        init_layer_k<<<gsz((long)NNODES * HID), BLK, 0, stream>>>(accum, m_u, denom);
        edge_max_k<<<gsz(EP), BLK, 0, stream>>>(ei, a_s, a_d, m_u);
        edge_sum_k<<<gsz(EP), BLK, 0, stream>>>(ei, a_s, a_d, m_u, denom);
        edge_msg_k<<<(int)EP, HID, 0, stream>>>(ei, a_s, a_d, m_u, denom, hW, accum);
        finalize_k<<<NNODES, HID, 0, stream>>>(accum, gat_b + (size_t)l * HID,
                                               ln_g + (size_t)l * HID, ln_b + (size_t)l * HID,
                                               h_, hb);
    }

    // head
    gather_k<<<gsz((long)NHYP * HID), BLK, 0, stream>>>(h_, hyp_idx, hyp_b);
    launch_gemm(hyp_b, head_W1p, head_b1, hyp_t, NHYP, 64, 1, stream);
    head2_k<<<gsz(NHYP), BLK, 0, stream>>>(hyp_t, head_W2, head_b2, logits);
}
